// Attention_82961588290112
// MI455X (gfx1250) — matrix operand order repaired and checked by reading, not compiled, not run on hardware
//
#include <hip/hip_runtime.h>

#define DEVINL __device__ __forceinline__

typedef __attribute__((ext_vector_type(16))) __bf16 v16bf;
typedef __attribute__((ext_vector_type(8)))  float  v8f;
typedef __attribute__((ext_vector_type(4)))  unsigned int u32x4;
typedef __attribute__((ext_vector_type(8)))  int i32x8;
typedef __attribute__((ext_vector_type(4)))  int i32x4;

union B16Tile { v16bf v; uint4 u4[2]; };

constexpr int B_  = 4;
constexpr int C_  = 64;    // channels
constexpr int HW_ = 4096;  // spatial positions
constexpr int CP_ = 32;    // channel-branch C/8=8, zero-padded to one WMMA K-step
constexpr int KB_ = 64;    // keys per flash block (two WMMA K-depths)

DEVINL v8f wmma_bf16(v16bf a, v16bf b, v8f c) {
  // emits v_wmma_f32_16x16x32_bf16
  return __builtin_amdgcn_wmma_f32_16x16x32_bf16(false, a, false, b, (short)0, c,
                                                 false, false);
}

DEVINL v8f vzero8() {
  v8f z;
#pragma unroll
  for (int i = 0; i < 8; ++i) z[i] = 0.f;
  return z;
}

// --------------------------------------------------------------------------
// TDM: 2D tile DMA global -> LDS. D# per CDNA5 ISA ch.8:
//  group0: count=1 | lds_addr | global_addr[56:0] | type=2
//  group1: data_size=3 (8B units), tensor dims huge (no OOB clipping),
//          tile_dim0 (units), tile_dim1 (rows), tensor_dim0_stride (units)
// Groups 2/3 unused (2D tile) -> zero. Tracked by TENSORcnt; EXEC ignored ->
// issue from one wave only.
// --------------------------------------------------------------------------
DEVINL void tdm_load_2d(unsigned int lds_off, const void* gptr,
                        unsigned int tile_d0, unsigned int tile_d1,
                        unsigned long long stride0_units)
{
  unsigned long long ga = (unsigned long long)(uintptr_t)gptr;
  u32x4 g0;
  g0[0] = 1u;                                            // count=1 user descriptor
  g0[1] = lds_off;                                       // LDS byte address
  g0[2] = (unsigned int)ga;                              // global addr lo
  g0[3] = (unsigned int)((ga >> 32) & 0x01FFFFFFu) | 0x80000000u; // hi25 | type=2
  const unsigned int td0 = 0x40000000u, td1 = 0x40000000u;
  i32x8 g1;
  g1[0] = (int)(3u << 16);                               // data_size = 8 bytes
  g1[1] = (int)((td0 & 0xFFFFu) << 16);                  // tensor_dim0 lo16
  g1[2] = (int)((td0 >> 16) | ((td1 & 0xFFFFu) << 16));  // dim0 hi16 | dim1 lo16
  g1[3] = (int)((td1 >> 16) | (tile_d0 << 16));          // dim1 hi16 | tile_dim0
  g1[4] = (int)(tile_d1 & 0xFFFFu);                      // tile_dim1 (tile_dim2=0)
  g1[5] = (int)(unsigned int)(stride0_units & 0xFFFFFFFFu);
  g1[6] = (int)(unsigned int)((stride0_units >> 32) & 0xFFFFu);
  g1[7] = 0;
  i32x4 z4; z4[0] = z4[1] = z4[2] = z4[3] = 0;
  i32x8 z8;
#pragma unroll
  for (int i = 0; i < 8; ++i) z8[i] = 0;
  // 6-arg form (clang-23 / therock-10.0 headers): groups 2,3 + spare group, cpol
  __builtin_amdgcn_tensor_load_to_lds(g0, g1, z4, z4, z8, 0);
}

DEVINL unsigned int lds_off_of(const void* p) {
  return (unsigned int)(uintptr_t)p;   // generic LDS addr: low 32 bits = LDS offset
}

// --------------------------------------------------------------------------
// Kernel 1: 1x1-conv projections -> bf16 staging buffers:
//   Qb  (B,HW,64) query-major, Kb (B,HW,64) key-major,
//   Vt  (B,64,HW) channel-major, Q1b/K1b (B,HW,32) zero-padded
// --------------------------------------------------------------------------
__global__ __launch_bounds__(256) void proj_kernel(
    const float* __restrict__ x, const float* __restrict__ y,
    const float* __restrict__ Wq, const float* __restrict__ bq,
    const float* __restrict__ Wk, const float* __restrict__ bk,
    const float* __restrict__ Wv, const float* __restrict__ bv,
    const float* __restrict__ Wq1, const float* __restrict__ bq1,
    const float* __restrict__ Wk1, const float* __restrict__ bk1,
    __bf16* __restrict__ Qb, __bf16* __restrict__ Kb, __bf16* __restrict__ Vt,
    __bf16* __restrict__ Q1b, __bf16* __restrict__ K1b)
{
  __shared__ float sWq[C_ * C_], sWk[C_ * C_], sWv[C_ * C_];
  __shared__ float sWq1[8 * C_], sWk1[8 * C_];
  __shared__ float sb[C_ * 3 + 16];

  const int tid = threadIdx.x;
  for (int i = tid; i < C_ * C_; i += 256) { sWq[i] = Wq[i]; sWk[i] = Wk[i]; sWv[i] = Wv[i]; }
  for (int i = tid; i < 8 * C_; i += 256) { sWq1[i] = Wq1[i]; sWk1[i] = Wk1[i]; }
  if (tid < C_) { sb[tid] = bq[tid]; sb[C_ + tid] = bk[tid]; sb[2 * C_ + tid] = bv[tid]; }
  if (tid < 8)  { sb[3 * C_ + tid] = bq1[tid]; sb[3 * C_ + 8 + tid] = bk1[tid]; }
  __syncthreads();

  const int b = blockIdx.x >> 4;
  const int q = ((blockIdx.x & 15) << 8) + tid;

  float col[C_];
  const float* xb = x + (size_t)b * C_ * HW_ + q;
#pragma unroll
  for (int c = 0; c < C_; ++c) col[c] = xb[(size_t)c * HW_];

  __bf16* krow = Kb + ((size_t)b * HW_ + q) * C_;
  __bf16* vcol = Vt + (size_t)b * C_ * HW_ + q;
  for (int oc = 0; oc < C_; ++oc) {
    float ak = sb[C_ + oc], av = sb[2 * C_ + oc];
    const float* wk = sWk + oc * C_;
    const float* wv = sWv + oc * C_;
#pragma unroll
    for (int ic = 0; ic < C_; ++ic) { ak += wk[ic] * col[ic]; av += wv[ic] * col[ic]; }
    krow[oc] = (__bf16)ak;
    vcol[(size_t)oc * HW_] = (__bf16)av;
  }

  __bf16* q1row = Q1b + ((size_t)b * HW_ + q) * CP_;
  __bf16* k1row = K1b + ((size_t)b * HW_ + q) * CP_;
  for (int oc = 0; oc < 8; ++oc) {
    float a1 = sb[3 * C_ + oc], k1 = sb[3 * C_ + 8 + oc];
    const float* w1 = sWq1 + oc * C_;
    const float* w2 = sWk1 + oc * C_;
#pragma unroll
    for (int ic = 0; ic < C_; ++ic) { a1 += w1[ic] * col[ic]; k1 += w2[ic] * col[ic]; }
    q1row[oc] = (__bf16)a1;
    k1row[oc] = (__bf16)k1;
  }
  for (int oc = 8; oc < CP_; ++oc) { q1row[oc] = (__bf16)0.f; k1row[oc] = (__bf16)0.f; }

  const float* yb = y + (size_t)b * C_ * HW_ + q;
#pragma unroll
  for (int c = 0; c < C_; ++c) col[c] = yb[(size_t)c * HW_];
  __bf16* qrow = Qb + ((size_t)b * HW_ + q) * C_;
  for (int oc = 0; oc < C_; ++oc) {
    float aq = sb[oc];
    const float* wq = sWq + oc * C_;
#pragma unroll
    for (int ic = 0; ic < C_; ++ic) aq += wq[ic] * col[ic];
    qrow[oc] = (__bf16)aq;
  }
}

// Online softmax over one 16q x 64k score block (four C/D tiles).
// C/D layout: VGPR r, lanes 0-15 -> row r, lanes 16-31 -> row r+8; col = lane&15.
// Row stats reduce across the 16-lane half (shfl_xor masks 1/2/4/8).
DEVINL void softmax_update4(v8f S[4], float m[8], float l[8], v8f O[4],
                            __bf16* pbuf, int ln, int half)
{
#pragma unroll
  for (int r = 0; r < 8; ++r) {
    float s0 = S[0][r], s1 = S[1][r], s2 = S[2][r], s3 = S[3][r];
    float mx = fmaxf(fmaxf(s0, s1), fmaxf(s2, s3));
#pragma unroll
    for (int s = 1; s <= 8; s <<= 1) mx = fmaxf(mx, __shfl_xor(mx, s, 32));
    float mn = fmaxf(m[r], mx);
    float p0 = __expf(s0 - mn);
    float p1 = __expf(s1 - mn);
    float p2 = __expf(s2 - mn);
    float p3 = __expf(s3 - mn);
    float rs = (p0 + p1) + (p2 + p3);
#pragma unroll
    for (int s = 1; s <= 8; s <<= 1) rs += __shfl_xor(rs, s, 32);
    float sc = __expf(m[r] - mn);
    l[r] = l[r] * sc + rs;
    m[r] = mn;
#pragma unroll
    for (int t = 0; t < 4; ++t) O[t][r] *= sc;
    const int row = r + half * 8;
    pbuf[row * KB_ + ln]      = (__bf16)p0;
    pbuf[row * KB_ + 16 + ln] = (__bf16)p1;
    pbuf[row * KB_ + 32 + ln] = (__bf16)p2;
    pbuf[row * KB_ + 48 + ln] = (__bf16)p3;
  }
}

// O += P @ V^T over a 64-key block: 2 A-tiles (P, keys 0-31 / 32-63),
// 4x2 B-tiles from channel-major V staged in LDS [64ch][64key].
DEVINL void pv_update(const __bf16* pbuf, const __bf16* vbuf, v8f O[4],
                      int ln, int half)
{
  B16Tile Ap0, Ap1;
  const __bf16* pr = pbuf + ln * KB_ + half * 16;
  Ap0.u4[0] = *(const uint4*)(pr);
  Ap0.u4[1] = *(const uint4*)(pr + 8);
  Ap1.u4[0] = *(const uint4*)(pr + 32);
  Ap1.u4[1] = *(const uint4*)(pr + 40);
#pragma unroll
  for (int t = 0; t < 4; ++t) {
    const __bf16* vr = vbuf + (t * 16 + ln) * KB_ + half * 16;
    B16Tile Bv0, Bv1;
    Bv0.u4[0] = *(const uint4*)(vr);
    Bv0.u4[1] = *(const uint4*)(vr + 8);
    Bv1.u4[0] = *(const uint4*)(vr + 32);
    Bv1.u4[1] = *(const uint4*)(vr + 40);
    O[t] = wmma_bf16(Ap0.v, Bv0.v, O[t]);
    O[t] = wmma_bf16(Ap1.v, Bv1.v, O[t]);
  }
}

// --------------------------------------------------------------------------
// Kernel 2: fused dual-softmax flash attention, TDM double-buffered tiles.
// Block = 128 threads (4 waves), wave owns 16 queries x 64 out channels.
// Grid = B * HW/64 = 256 blocks. 28 WMMA per 64-key block per wave.
// --------------------------------------------------------------------------
__global__ __launch_bounds__(128) void flash_attn_kernel(
    const __bf16* __restrict__ Qb, const __bf16* __restrict__ Kb,
    const __bf16* __restrict__ Vt, const __bf16* __restrict__ Q1b,
    const __bf16* __restrict__ K1b, float* __restrict__ out)
{
  __shared__ alignas(16) __bf16 ldsK [2][KB_ * C_];    // [key][ch]     8KB each
  __shared__ alignas(16) __bf16 ldsV [2][C_ * KB_];    // [ch][key]     8KB each
  __shared__ alignas(16) __bf16 ldsK1[2][KB_ * CP_];   // [key][ch-pad] 4KB each
  __shared__ alignas(16) __bf16 ldsP [4][16 * KB_];    // per-wave P    2KB each

  const int tid  = threadIdx.x;
  const int wave = tid >> 5;
  const int lane = tid & 31;
  const int half = lane >> 4;
  const int ln   = lane & 15;

  const int b     = blockIdx.x >> 6;
  const int qbase = (blockIdx.x & 63) * 64 + wave * 16;

  const __bf16* Qb_b  = Qb  + (size_t)b * HW_ * C_;
  const __bf16* Kb_b  = Kb  + (size_t)b * HW_ * C_;
  const __bf16* Vt_b  = Vt  + (size_t)b * C_ * HW_;
  const __bf16* Q1b_b = Q1b + (size_t)b * HW_ * CP_;
  const __bf16* K1b_b = K1b + (size_t)b * HW_ * CP_;

  // Q A-tiles: lane ln holds query row qbase+ln, K chunk {h*16..h*16+15}.
  const __bf16* qrow = Qb_b + (size_t)(qbase + ln) * C_ + half * 16;
  B16Tile Aq0, Aq1, Aq2;
  Aq0.u4[0] = *(const uint4*)(qrow);
  Aq0.u4[1] = *(const uint4*)(qrow + 8);
  Aq1.u4[0] = *(const uint4*)(qrow + 32);
  Aq1.u4[1] = *(const uint4*)(qrow + 40);
  const __bf16* q1row = Q1b_b + (size_t)(qbase + ln) * CP_ + half * 16;
  Aq2.u4[0] = *(const uint4*)(q1row);
  Aq2.u4[1] = *(const uint4*)(q1row + 8);

  v8f O0[4], O1[4];
  float m0[8], l0[8], m1[8], l1[8];
#pragma unroll
  for (int t = 0; t < 4; ++t) {
#pragma unroll
    for (int r = 0; r < 8; ++r) { O0[t][r] = 0.f; O1[t][r] = 0.f; }
  }
#pragma unroll
  for (int r = 0; r < 8; ++r) { m0[r] = -1e30f; m1[r] = -1e30f; l0[r] = 0.f; l1[r] = 0.f; }

  // TDM stage of one 64-key tile set into buffer `buf` (wave 0 only issues).
  auto stage = [&](int buf, int kb) {
    // K tile: 64 keys x 128B, fully contiguous -> 1 row of 1024 x 8B units
    tdm_load_2d(lds_off_of(&ldsK[buf][0]), Kb_b + (size_t)kb * C_,
                KB_ * C_ * 2 / 8, 1, KB_ * C_ * 2 / 8);
    // V tile: 64 channel rows x 128B, row stride HW*2 bytes = 1024 units
    tdm_load_2d(lds_off_of(&ldsV[buf][0]), Vt_b + kb,
                KB_ * 2 / 8, C_, (unsigned long long)(HW_ * 2 / 8));
    // K1 tile: contiguous 4KB -> 1 row of 512 x 8B units
    tdm_load_2d(lds_off_of(&ldsK1[buf][0]), K1b_b + (size_t)kb * CP_,
                KB_ * CP_ * 2 / 8, 1, KB_ * CP_ * 2 / 8);
  };

  if (wave == 0) {
    stage(0, 0);
    __builtin_amdgcn_s_wait_tensorcnt(0);
  }
  __syncthreads();

  int cur = 0;
  for (int kb = 0; kb < HW_; kb += KB_) {
    const bool more = (kb + KB_) < HW_;
    if (more && wave == 0) stage(cur ^ 1, kb + KB_);   // async prefetch next tile

    const __bf16* Kc  = ldsK [cur];
    const __bf16* Vc  = ldsV [cur];
    const __bf16* K1c = ldsK1[cur];
    __bf16* pbuf = ldsP[wave];

    // ---- branch 0: S = Q^T K (16q x 64k), K-depth 64 = 2 WMMA per tile ----
    {
      v8f S[4];
#pragma unroll
      for (int n = 0; n < 4; ++n) {
        const __bf16* kr = Kc + (n * 16 + ln) * C_ + half * 16;
        B16Tile Bk0, Bk1;
        Bk0.u4[0] = *(const uint4*)(kr);
        Bk0.u4[1] = *(const uint4*)(kr + 8);
        Bk1.u4[0] = *(const uint4*)(kr + 32);
        Bk1.u4[1] = *(const uint4*)(kr + 40);
        v8f acc = vzero8();
        acc = wmma_bf16(Aq0.v, Bk0.v, acc);
        acc = wmma_bf16(Aq1.v, Bk1.v, acc);
        S[n] = acc;
      }
      softmax_update4(S, m0, l0, O0, pbuf, ln, half);
      pv_update(pbuf, Vc, O0, ln, half);
    }

    // ---- branch 1: T = Q1^T K1 (padded K-depth 32 = 1 WMMA per tile) ----
    {
      v8f T[4];
#pragma unroll
      for (int n = 0; n < 4; ++n) {
        const __bf16* k1r = K1c + (n * 16 + ln) * CP_ + half * 16;
        B16Tile Bq;
        Bq.u4[0] = *(const uint4*)(k1r);
        Bq.u4[1] = *(const uint4*)(k1r + 8);
        T[n] = wmma_bf16(Aq2.v, Bq.v, vzero8());
      }
      softmax_update4(T, m1, l1, O1, pbuf, ln, half);
      pv_update(pbuf, Vc, O1, ln, half);
    }

    if (more && wave == 0) __builtin_amdgcn_s_wait_tensorcnt(0);
    __syncthreads();   // prefetch landed + everyone done with `cur`
    cur ^= 1;
  }

  // out[b][c][q] = O0/l0 + O1/l1
  const size_t ob = (size_t)b * C_ * HW_;
#pragma unroll
  for (int t = 0; t < 4; ++t) {
    const int c = t * 16 + ln;
#pragma unroll
    for (int r = 0; r < 8; ++r) {
      const int q = qbase + r + half * 8;
      out[ob + (size_t)c * HW_ + q] = O0[t][r] / l0[r] + O1[t][r] / l1[r];
    }
  }
}

// --------------------------------------------------------------------------
extern "C" void kernel_launch(void* const* d_in, const int* in_sizes, int n_in,
                              void* d_out, int out_size, void* d_ws, size_t ws_size,
                              hipStream_t stream) {
  (void)in_sizes; (void)n_in; (void)out_size; (void)ws_size;
  const float* x   = (const float*)d_in[0];
  const float* y   = (const float*)d_in[1];
  const float* Wq  = (const float*)d_in[2];
  const float* bq  = (const float*)d_in[3];
  const float* Wk  = (const float*)d_in[4];
  const float* bk  = (const float*)d_in[5];
  const float* Wv  = (const float*)d_in[6];
  const float* bv  = (const float*)d_in[7];
  const float* Wq1 = (const float*)d_in[8];
  const float* bq1 = (const float*)d_in[9];
  const float* Wk1 = (const float*)d_in[10];
  const float* bk1 = (const float*)d_in[11];

  char* ws = (char*)d_ws;                      // 8 MB of bf16 staging
  __bf16* Qb  = (__bf16*)(ws + (size_t)0);
  __bf16* Kb  = (__bf16*)(ws + (size_t)2 * 1024 * 1024);
  __bf16* Vt  = (__bf16*)(ws + (size_t)4 * 1024 * 1024);
  __bf16* Q1b = (__bf16*)(ws + (size_t)6 * 1024 * 1024);
  __bf16* K1b = (__bf16*)(ws + (size_t)7 * 1024 * 1024);

  hipLaunchKernelGGL(proj_kernel, dim3(B_ * 16), dim3(256), 0, stream,
                     x, y, Wq, bq, Wk, bk, Wv, bv, Wq1, bq1, Wk1, bk1,
                     Qb, Kb, Vt, Q1b, K1b);
  hipLaunchKernelGGL(flash_attn_kernel, dim3(B_ * (HW_ / 64)), dim3(128), 0, stream,
                     Qb, Kb, Vt, Q1b, K1b, (float*)d_out);
}
